// TTTLayer_76879914598613
// MI455X (gfx1250) — compile-verified, matrix operand order checked by reading, not run on hardware
//
#include <hip/hip_runtime.h>
#include <hip/hip_bf16.h>
#include <cstdint>

typedef unsigned short u16;
typedef __attribute__((ext_vector_type(16))) __bf16 bfrag_t;
typedef __attribute__((ext_vector_type(8)))  float  v8f;

typedef __attribute__((ext_vector_type(4))) unsigned int tdm_g0_t;
typedef __attribute__((ext_vector_type(8))) int          tdm_g1_t;
typedef __attribute__((ext_vector_type(4))) int          tdm_g23_t;

struct U8x { uint4 a, b; };

__device__ __forceinline__ u16 f2bf(float f) {
    unsigned u = __builtin_bit_cast(unsigned, f);
    u += 0x7fffu + ((u >> 16) & 1u);   // round-to-nearest-even
    return (u16)(u >> 16);
}

struct EpiP {
    void* p0; void* p1; void* p2;
    long long zs0;      // z-stride (elements) for p0
    int ldc;
    float scale;
    float* partials;
    int pbase;
    int aux;            // chunk base for o-store
};

#define BM 128
#define BN 128
#define BKK 32
#define LDK 40          // BKK + 8 pad -> 80B row stride (16B aligned, conflict-reduced)

enum { MODE_NN = 0, MODE_TN = 1, MODE_NT = 2 };
enum { EPI_F32 = 0, EPI_BF16 = 1, EPI_QKV = 2, EPI_DPRED = 3, EPI_OSTORE = 4, EPI_WUPD = 5 };

#define LR_C 0.001f

#if __has_builtin(__builtin_amdgcn_tensor_load_to_lds)
#define HAVE_TDM 1
#else
#define HAVE_TDM 0
#endif

#if HAVE_TDM
// ---------------------------------------------------------------------------
// TDM: DMA a 128-row x 32-element (bf16) 2D tile from global into LDS with
// hardware row padding: 16 DWORDs of data (pad_interval code 3) followed by
// 4 DWORDs of pad (pad_amount code 3)  ->  80-byte LDS row stride == LDK.
// Descriptor per CDNA5 ISA (D# group0/group1, type=2).
// amdgpu-toolchain (clang-23) builtin form: 6 args (g0, g1, g2, g3, g4, cpol).
// ---------------------------------------------------------------------------
__device__ __forceinline__ void tdm_load_tile_128x32(unsigned lds_off,
                                                     const u16* gsrc,
                                                     int stride_elems) {
    const unsigned long long ga = (unsigned long long)(const void*)gsrc;
    tdm_g0_t g0 = { 1u,                                   // count=1 (valid user D#)
                    lds_off,                              // lds_addr (bytes)
                    (unsigned)ga,                         // global_addr[31:0]
                    (unsigned)((ga >> 32) & 0x01ffffffu)  // global_addr[56:32]
                        | (2u << 30) };                   // type=2 ("image")
    const unsigned td0 = 0x40000000u, td1 = 0x40000000u;  // huge dims: no OOB clamp
    tdm_g1_t g1 = {
        (int)((1u << 16)            // data_size = 2 bytes
              | (1u << 20)          // pad_enable
              | (3u << 22)          // pad_interval: 16 DWORDs (64B = one 32-elem row)
              | (3u << 25)),        // pad_amount: 4 DWORDs (16B = 8 elems)
        (int)((td0 & 0xffffu) << 16),                               // tensor_dim0[15:0]
        (int)(((td0 >> 16) & 0xffffu) | ((td1 & 0xffffu) << 16)),   // td0[31:16]|td1[15:0]
        (int)(((td1 >> 16) & 0xffffu) | (32u << 16)),               // td1[31:16]|tile_dim0=32
        (int)128u,                                                  // tile_dim1=128, tile_dim2=0
        (int)stride_elems,                                          // tensor_dim0_stride[31:0]
        0,                                                          // stride[47:32] | td1_stride lo
        0 };
    tdm_g23_t z4 = { 0, 0, 0, 0 };
    tdm_g1_t  z8 = { 0, 0, 0, 0, 0, 0, 0, 0 };
    __builtin_amdgcn_tensor_load_to_lds(g0, g1, z4, z4, z8, 0);
}
#endif

// ---------------------------------------------------------------------------
// Generic bf16 WMMA GEMM. C[M,N] over grid (N/BN, M/BM, Z).
//   MODE_NN: C = A[M,K] * B[K,N]
//   MODE_TN: C = A[K,M]^T * B[K,N]
//   MODE_NT: C = A[M,K] * B[N,K]^T
// 8 waves per block, wave grid 2(M) x 4(N), wave tile 64x32 (4x2 WMMA tiles).
// Non-transposing tile copies go through the Tensor Data Mover (async DMA,
// TENSORcnt); transposing copies ([K,N] -> Bs[n][k]) stay on the VGPR path.
// ---------------------------------------------------------------------------
template<int MODE, int EPI>
__global__ void __launch_bounds__(256)
gemm_bf16(const u16* __restrict__ A, const u16* __restrict__ B,
          int K, int lda, int ldb, long long zsA, long long zsB, EpiP ep)
{
    __shared__ u16 As[BM][LDK];
    __shared__ u16 Bs[BN][LDK];   // always stored [n][k]

    const int tid  = threadIdx.x;
    const int lane = tid & 31;
    const int wv   = tid >> 5;
    const int wm   = wv >> 2;       // 0..1
    const int wn   = wv & 3;        // 0..3
    const int lm   = lane & 15;
    const int k0   = (lane >> 4) * 8;   // 0 or 8 (K-offset of fragment half)

    const int m0 = blockIdx.y * BM;
    const int n0 = blockIdx.x * BN;
    const int z  = blockIdx.z;

    A += (long long)z * zsA;
    B += (long long)z * zsB;

#if HAVE_TDM
    const unsigned asOff = (unsigned)(size_t)(void*)&As[0][0];
    const unsigned bsOff = (unsigned)(size_t)(void*)&Bs[0][0];
#endif

    v8f acc[4][2] = {};

    for (int kk = 0; kk < K; kk += BKK) {
        // ---- stage A tile -> As[m][k] ----
        if (MODE == MODE_NN || MODE == MODE_NT) {
#if HAVE_TDM
            if (wv == 0)
                tdm_load_tile_128x32(asOff, A + (long long)m0 * lda + kk, lda);
#else
            const int kc = tid & 3, r = tid >> 2;            // r: 0..63
            #pragma unroll
            for (int h = 0; h < 2; h++) {
                const int row = r + 64 * h;
                uint4 v = *(const uint4*)(A + (long long)(m0 + row) * lda + kk + kc * 8);
                *(uint4*)&As[row][kc * 8] = v;
            }
#endif
        } else {  // TN: A stored [K, M] -> transpose into As
            const int mc = tid & 15, kr = tid >> 4;          // kr: 0..15
            #pragma unroll
            for (int h = 0; h < 2; h++) {
                const int krow = kr + 16 * h;
                uint4 v = *(const uint4*)(A + (long long)(kk + krow) * lda + m0 + mc * 8);
                u16 e[8] = { (u16)(v.x & 0xffff), (u16)(v.x >> 16),
                             (u16)(v.y & 0xffff), (u16)(v.y >> 16),
                             (u16)(v.z & 0xffff), (u16)(v.z >> 16),
                             (u16)(v.w & 0xffff), (u16)(v.w >> 16) };
                #pragma unroll
                for (int j = 0; j < 8; j++) As[mc * 8 + j][krow] = e[j];
            }
        }
        // ---- stage B tile -> Bs[n][k] ----
        if (MODE == MODE_NT) {  // B stored [N, K] -> direct copy (DMA-able)
#if HAVE_TDM
            if (wv == 0)
                tdm_load_tile_128x32(bsOff, B + (long long)n0 * ldb + kk, ldb);
#else
            const int kc = tid & 3, r = tid >> 2;
            #pragma unroll
            for (int h = 0; h < 2; h++) {
                const int row = r + 64 * h;
                uint4 v = *(const uint4*)(B + (long long)(n0 + row) * ldb + kk + kc * 8);
                *(uint4*)&Bs[row][kc * 8] = v;
            }
#endif
        } else {  // NN/TN: B stored [K, N] -> transpose
            const int nc = tid & 15, kr = tid >> 4;
            #pragma unroll
            for (int h = 0; h < 2; h++) {
                const int krow = kr + 16 * h;
                uint4 v = *(const uint4*)(B + (long long)(kk + krow) * ldb + n0 + nc * 8);
                u16 e[8] = { (u16)(v.x & 0xffff), (u16)(v.x >> 16),
                             (u16)(v.y & 0xffff), (u16)(v.y >> 16),
                             (u16)(v.z & 0xffff), (u16)(v.z >> 16),
                             (u16)(v.w & 0xffff), (u16)(v.w >> 16) };
                #pragma unroll
                for (int j = 0; j < 8; j++) Bs[nc * 8 + j][krow] = e[j];
            }
        }
#if HAVE_TDM
        if ((MODE == MODE_NN || MODE == MODE_NT) && wv == 0)
            __builtin_amdgcn_s_wait_tensorcnt(0);
#endif
        __syncthreads();

        // ---- fragments: per ISA 16-bit A layout, e<8 -> K=k0+e, e>=8 -> K=16+k0+(e-8)
        bfrag_t af[4], bfr[2];
        #pragma unroll
        for (int mt = 0; mt < 4; mt++) {
            const u16* p = &As[wm * 64 + mt * 16 + lm][0];
            U8x t;
            t.a = *(const uint4*)(p + k0);
            t.b = *(const uint4*)(p + k0 + 16);
            af[mt] = __builtin_bit_cast(bfrag_t, t);
        }
        #pragma unroll
        for (int nt = 0; nt < 2; nt++) {
            const u16* p = &Bs[wn * 32 + nt * 16 + lm][0];
            U8x t;
            t.a = *(const uint4*)(p + k0);
            t.b = *(const uint4*)(p + k0 + 16);
            bfr[nt] = __builtin_bit_cast(bfrag_t, t);
        }
        #pragma unroll
        for (int mt = 0; mt < 4; mt++)
            #pragma unroll
            for (int nt = 0; nt < 2; nt++)
                acc[mt][nt] = __builtin_amdgcn_wmma_f32_16x16x32_bf16(
                    false, af[mt], false, bfr[nt], (short)0, acc[mt][nt], false, false);
        __syncthreads();
    }

    // ---- epilogue (C/D layout: lane L, vgpr r -> row = r + 8*(L>=16), col = L%16)
    float lsum = 0.f;
    const int lh = lane >> 4;
    #pragma unroll
    for (int mt = 0; mt < 4; mt++) {
        #pragma unroll
        for (int nt = 0; nt < 2; nt++) {
            #pragma unroll
            for (int vr = 0; vr < 8; vr++) {
                const int row = m0 + wm * 64 + mt * 16 + vr + 8 * lh;
                const int col = n0 + wn * 32 + nt * 16 + lm;
                const float val = acc[mt][nt][vr];
                if constexpr (EPI == EPI_F32) {
                    float* C = (float*)ep.p0 + (long long)z * ep.zs0;
                    C[(long long)row * ep.ldc + col] = val;
                } else if constexpr (EPI == EPI_BF16) {
                    u16* C = (u16*)ep.p0 + (long long)z * ep.zs0;
                    C[(long long)row * ep.ldc + col] = f2bf(val);
                } else if constexpr (EPI == EPI_QKV) {
                    const int b = row >> 11, s = row & 2047;
                    if (col < 768) {
                        ((u16*)ep.p0)[(long long)row * 768 + col] = f2bf(val);
                    } else {
                        // chunk-permuted [NC, B, C, H] layout for k/v
                        const long long pr =
                            (long long)((s >> 9) * 2048 + (b << 9) + (s & 511)) * 768;
                        if (col < 1536) ((u16*)ep.p1)[pr + (col - 768)] = f2bf(val);
                        else            ((float*)ep.p2)[pr + (col - 1536)] = val;
                    }
                } else if constexpr (EPI == EPI_DPRED) {
                    const long long idx = (long long)row * ep.ldc + col;
                    const float diff = val - ((const float*)ep.p0)[idx];
                    ((u16*)ep.p1)[idx] = f2bf(diff * ep.scale);
                    lsum += diff * diff;
                } else if constexpr (EPI == EPI_OSTORE) {
                    const int b = row >> 9, s = ep.aux + (row & 511);
                    ((u16*)ep.p0)[((long long)(b * 2048 + s)) * 768 + col] = f2bf(val);
                } else if constexpr (EPI == EPI_WUPD) {
                    const long long idx = (long long)row * ep.ldc + col;
                    const float nv = ((float*)ep.p0)[idx] - ep.scale * val;
                    ((float*)ep.p0)[idx] = nv;
                    ((u16*)ep.p1)[idx] = f2bf(nv);
                }
            }
        }
    }
    if constexpr (EPI == EPI_DPRED) {
        __syncthreads();
        float* red = (float*)&As[0][0];
        red[tid] = lsum;
        __syncthreads();
        for (int s = 128; s > 0; s >>= 1) {
            if (tid < s) red[tid] += red[tid + s];
            __syncthreads();
        }
        if (tid == 0)
            ep.partials[ep.pbase + blockIdx.y * gridDim.x + blockIdx.x] = red[0];
    }
}

// ---------------------------------------------------------------------------
// Elementwise helpers
// ---------------------------------------------------------------------------
__global__ void k_f32_to_bf16(const float* __restrict__ in, u16* __restrict__ out, long long n) {
    long long i = (long long)blockIdx.x * blockDim.x + threadIdx.x;
    const long long st = (long long)gridDim.x * blockDim.x;
    for (; i < n; i += st) out[i] = f2bf(in[i]);
}

__global__ void k_init_w(const float* __restrict__ src, float* __restrict__ wf,
                         u16* __restrict__ wh, long long n) {
    long long i = (long long)blockIdx.x * blockDim.x + threadIdx.x;
    const long long st = (long long)gridDim.x * blockDim.x;
    for (; i < n; i += st) { const float v = src[i]; wf[i] = v; wh[i] = f2bf(v); }
}

__global__ void k_fuseG(const float* __restrict__ Z0, const float* __restrict__ Z2,
                        u16* __restrict__ G, long long n) {
    long long i = (long long)blockIdx.x * blockDim.x + threadIdx.x;
    const long long st = (long long)gridDim.x * blockDim.x;
    for (; i < n; i += st) {
        const float z0 = Z0[i];
        const float sg = 1.f / (1.f + __expf(-z0));
        G[i] = f2bf(z0 * sg * Z2[i]);
    }
}

__global__ void k_dZ(const float* __restrict__ Z0, const float* __restrict__ Z2,
                     const float* __restrict__ dG,
                     u16* __restrict__ dZ0, u16* __restrict__ dZ2, long long n) {
    long long i = (long long)blockIdx.x * blockDim.x + threadIdx.x;
    const long long st = (long long)gridDim.x * blockDim.x;
    for (; i < n; i += st) {
        const float z0 = Z0[i], z2 = Z2[i], g = dG[i];
        const float sg = 1.f / (1.f + __expf(-z0));
        const float a  = z0 * sg;                       // silu(z0)
        const float ds = sg * (1.f + z0 * (1.f - sg));  // silu'(z0)
        dZ0[i] = f2bf(g * z2 * ds);
        dZ2[i] = f2bf(g * a);
    }
}

__global__ void k_loss(const float* __restrict__ partials, float* __restrict__ out, float inv) {
    __shared__ float red[256];
    float s = 0.f;
    for (int i = threadIdx.x; i < 512; i += 256)
        if ((i & 127) < 96) s += partials[i];
    red[threadIdx.x] = s;
    __syncthreads();
    for (int t = 128; t > 0; t >>= 1) {
        if ((int)threadIdx.x < t) red[threadIdx.x] += red[threadIdx.x + t];
        __syncthreads();
    }
    if (threadIdx.x == 0) out[0] = red[0] * inv;
}

// ---------------------------------------------------------------------------
extern "C" void kernel_launch(void* const* d_in, const int* in_sizes, int n_in,
                              void* d_out, int out_size, void* d_ws, size_t ws_size,
                              hipStream_t stream)
{
    (void)in_sizes; (void)n_in; (void)ws_size;
    const float* x    = (const float*)d_in[0];
    const float* Wqkv = (const float*)d_in[1];
    const float* w0   = (const float*)d_in[2];
    const float* w1   = (const float*)d_in[3];
    const float* w2   = (const float*)d_in[4];
    const float* Wout = (const float*)d_in[5];
    float* out = (float*)d_out;

    const int Bb = 4, S = 2048, H = 768, TT = 2048, C = 512;
    const long long BS = (long long)Bb * S;   // 8192 rows
    const long long BC = (long long)Bb * C;   // 2048 rows per chunk

    char* wp = (char*)d_ws;
    auto alloc = [&](long long bytes) -> void* {
        void* p = (void*)wp; wp += (bytes + 255) & ~255LL; return p;
    };

    u16*   xh      = (u16*)  alloc(BS * H * 2);
    u16*   Wqkvh   = (u16*)  alloc((long long)H * 3 * H * 2);
    u16*   qh      = (u16*)  alloc(BS * H * 2);
    u16*   kph     = (u16*)  alloc(BS * H * 2);      // [NC,B,C,H] permuted
    float* vpf     = (float*)alloc(BS * H * 4);      // [NC,B,C,H] permuted
    u16*   Wouth   = (u16*)  alloc((long long)S * H * 2);
    float* w0f     = (float*)alloc((long long)H * TT * 4);
    float* w1f     = (float*)alloc((long long)TT * H * 4);
    float* w2f     = (float*)alloc((long long)H * TT * 4);
    u16*   w0h     = (u16*)  alloc((long long)H * TT * 2);
    u16*   w1h     = (u16*)  alloc((long long)TT * H * 2);
    u16*   w2h     = (u16*)  alloc((long long)H * TT * 2);
    float* Z0      = (float*)alloc(BC * TT * 4);
    float* Z2      = (float*)alloc(BC * TT * 4);
    u16*   Gh      = (u16*)  alloc(BC * TT * 2);
    u16*   dpredh  = (u16*)  alloc(BC * H * 2);
    float* dGf     = (float*)alloc(BC * TT * 4);
    u16*   dZ0h    = (u16*)  alloc(BC * TT * 2);
    u16*   dZ2h    = (u16*)  alloc(BC * TT * 2);
    u16*   oh      = (u16*)  alloc(BS * H * 2);      // [B,S,H]
    u16*   Th      = (u16*)  alloc((long long)Bb * H * H * 2);
    float* parts   = (float*)alloc(512 * 4);

    const dim3 blk(256);

    // --- convert inputs to bf16 / init weight state ---
    k_f32_to_bf16<<<dim3(512), blk, 0, stream>>>(x, xh, BS * H);
    k_f32_to_bf16<<<dim3(256), blk, 0, stream>>>(Wqkv, Wqkvh, (long long)H * 3 * H);
    k_f32_to_bf16<<<dim3(256), blk, 0, stream>>>(Wout, Wouth, (long long)S * H);
    k_init_w<<<dim3(256), blk, 0, stream>>>(w0, w0f, w0h, (long long)H * TT);
    k_init_w<<<dim3(256), blk, 0, stream>>>(w1, w1f, w1h, (long long)TT * H);
    k_init_w<<<dim3(256), blk, 0, stream>>>(w2, w2f, w2h, (long long)H * TT);

    EpiP ep{};

    // --- qkv = x @ Wqkv, fused split into q / k_perm / v_perm ---
    ep = {}; ep.p0 = qh; ep.p1 = kph; ep.p2 = vpf;
    gemm_bf16<MODE_NN, EPI_QKV><<<dim3(3 * H / BN, BS / BM, 1), blk, 0, stream>>>(
        xh, Wqkvh, H, H, 3 * H, 0, 0, ep);

    const float Nel = (float)(BC * H);   // 4*512*768

    for (int ch = 0; ch < 4; ch++) {
        const u16*   kc = kph + (long long)ch * BC * H;
        const float* vc = vpf + (long long)ch * BC * H;

        // Z0 = Kc @ w0, Z2 = Kc @ w2
        ep = {}; ep.p0 = Z0; ep.ldc = TT;
        gemm_bf16<MODE_NN, EPI_F32><<<dim3(TT / BN, BC / BM), blk, 0, stream>>>(
            kc, w0h, H, H, TT, 0, 0, ep);
        ep.p0 = Z2;
        gemm_bf16<MODE_NN, EPI_F32><<<dim3(TT / BN, BC / BM), blk, 0, stream>>>(
            kc, w2h, H, H, TT, 0, 0, ep);
        k_fuseG<<<dim3(1024), blk, 0, stream>>>(Z0, Z2, Gh, BC * TT);

        // pred = G @ w1 ; fused dpred = 2*(pred - v)/N + loss partials
        ep = {}; ep.p0 = (void*)vc; ep.p1 = dpredh; ep.ldc = H;
        ep.scale = 2.f / Nel; ep.partials = parts; ep.pbase = ch * 128;
        gemm_bf16<MODE_NN, EPI_DPRED><<<dim3(H / BN, BC / BM), blk, 0, stream>>>(
            Gh, w1h, TT, TT, H, 0, 0, ep);

        // dG = dpred @ w1^T
        ep = {}; ep.p0 = dGf; ep.ldc = TT;
        gemm_bf16<MODE_NT, EPI_F32><<<dim3(TT / BN, BC / BM), blk, 0, stream>>>(
            dpredh, w1h, H, H, H, 0, 0, ep);

        // dw1 = G^T @ dpred ; fused w1 -= LR*dw1
        ep = {}; ep.p0 = w1f; ep.p1 = w1h; ep.ldc = H; ep.scale = LR_C;
        gemm_bf16<MODE_TN, EPI_WUPD><<<dim3(H / BN, TT / BM), blk, 0, stream>>>(
            Gh, dpredh, (int)BC, TT, H, 0, 0, ep);

        // dZ0, dZ2 from dG, Z0, Z2
        k_dZ<<<dim3(1024), blk, 0, stream>>>(Z0, Z2, dGf, dZ0h, dZ2h, BC * TT);

        // dw0 = Kc^T @ dZ0 ; w0 update
        ep = {}; ep.p0 = w0f; ep.p1 = w0h; ep.ldc = TT; ep.scale = LR_C;
        gemm_bf16<MODE_TN, EPI_WUPD><<<dim3(TT / BN, H / BM), blk, 0, stream>>>(
            kc, dZ0h, (int)BC, H, TT, 0, 0, ep);
        // dw2 = Kc^T @ dZ2 ; w2 update
        ep.p0 = w2f; ep.p1 = w2h;
        gemm_bf16<MODE_TN, EPI_WUPD><<<dim3(TT / BN, H / BM), blk, 0, stream>>>(
            kc, dZ2h, (int)BC, H, TT, 0, 0, ep);

        // forward with updated weights
        ep = {}; ep.p0 = Z0; ep.ldc = TT;
        gemm_bf16<MODE_NN, EPI_F32><<<dim3(TT / BN, BC / BM), blk, 0, stream>>>(
            kc, w0h, H, H, TT, 0, 0, ep);
        ep.p0 = Z2;
        gemm_bf16<MODE_NN, EPI_F32><<<dim3(TT / BN, BC / BM), blk, 0, stream>>>(
            kc, w2h, H, H, TT, 0, 0, ep);
        k_fuseG<<<dim3(1024), blk, 0, stream>>>(Z0, Z2, Gh, BC * TT);

        // o_chunk = Gn @ w1n, stored into o[B,S,H] with chunk remap
        ep = {}; ep.p0 = oh; ep.aux = ch * C;
        gemm_bf16<MODE_NN, EPI_OSTORE><<<dim3(H / BN, BC / BM), blk, 0, stream>>>(
            Gh, w1h, TT, TT, H, 0, 0, ep);
    }

    // --- reassociated attention: T[b] = o[b]^T @ Wout  (768x768, batched z=4) ---
    ep = {}; ep.p0 = Th; ep.ldc = H; ep.zs0 = (long long)H * H;
    gemm_bf16<MODE_TN, EPI_BF16><<<dim3(H / BN, H / BM, 4), blk, 0, stream>>>(
        oh, Wouth, S, H, H, (long long)S * H, 0, ep);

    // --- output[b] = q[b] @ T[b] -> d_out f32 ---
    ep = {}; ep.p0 = out; ep.ldc = H; ep.zs0 = (long long)S * H;
    gemm_bf16<MODE_NN, EPI_F32><<<dim3(H / BN, S / BM, 4), blk, 0, stream>>>(
        qh, Th, H, H, H, (long long)S * H, (long long)H * H, ep);

    // --- mean chunk loss: sum(diff^2)/(Nel) averaged over 4 chunks ---
    k_loss<<<dim3(1), blk, 0, stream>>>(parts, out + (out_size - 1), 1.f / (Nel * 4.f));
}
